// PostNormMamba2_68564857913701
// MI455X (gfx1250) — matrix operand order repaired and checked by reading, not compiled, not run on hardware
//
#include <hip/hip_runtime.h>
#include <hip/hip_bf16.h>

// ---------------------------------------------------------------------------
// PostNorm Mamba2 block for MI455X (gfx1250, wave32, WMMA).
//
// Pipeline:
//   prep   : x -> bf16 ; W_in -> W_in^T bf16 ; W_out -> W_out^T bf16
//   gemm1  : zxbcdt = x @ W_in       (v_wmma_f32_16x16x32_bf16, 32x32/wave,
//                                     ping-pong double-buffered b128 loads)
//   conv   : xBC = silu(causal depthwise conv)
//   scan   : selective state scan (softplus fused), 256 WGs
//   gate   : y = rms(y * silu(z)) * norm_w  -> bf16
//   gemm2  : og = y @ W_out          (same WMMA kernel)
//   final  : out = rms(og) + x
// ---------------------------------------------------------------------------

#define D_MODEL  1024
#define D_INNER  2048
#define NHEADS   32
#define HEADDIM  64
#define D_STATE  128
#define D_CONV   4
#define CONV_DIM (D_INNER + 2 * D_STATE)                 // 2304
#define DPROJ    (2 * D_INNER + 2 * D_STATE + NHEADS)    // 4384
#define BATCH    2
#define SEQ      2048
#define MROWS    (BATCH * SEQ)                           // 4096
#define EPSV     1e-5f
#define PSPLIT   4                                       // HEADDIM split for scan

typedef __attribute__((ext_vector_type(16))) __bf16 v16bf;
typedef __attribute__((ext_vector_type(8)))  float  v8f;

union V16 { v16bf v; uint4 u[2]; };

// --------------------------- prep kernels ----------------------------------

__global__ __launch_bounds__(256) void cvt_bf16_kernel(const float* __restrict__ src,
                                                       __bf16* __restrict__ dst, int n) {
  int i = blockIdx.x * 256 + threadIdx.x;
  if (i < n) dst[i] = (__bf16)src[i];
}

// src: R x C fp32 row-major  ->  dst: C x R bf16 row-major (transpose + convert)
__global__ __launch_bounds__(256) void transpose_cvt_kernel(const float* __restrict__ src,
                                                            __bf16* __restrict__ dst,
                                                            int R, int C) {
  __shared__ float tile[32][33];
  const int c0 = blockIdx.x * 32, r0 = blockIdx.y * 32;
  const int tx = threadIdx.x & 31, ty = threadIdx.x >> 5;  // 8 rows per sweep
#pragma unroll
  for (int i = 0; i < 32; i += 8)
    tile[ty + i][tx] = src[(size_t)(r0 + ty + i) * C + (c0 + tx)];
  __syncthreads();
#pragma unroll
  for (int i = 0; i < 32; i += 8)
    dst[(size_t)(c0 + ty + i) * R + (r0 + tx)] = (__bf16)tile[tx][ty + i];
}

// --------------------------- WMMA GEMM -------------------------------------
// C[M,N] (fp32) = A[M,K] (bf16, row-major) x Bt[N,K] (bf16 = B transposed).
// One wave -> 32x32 register tile (2 M-tiles x 2 N-tiles, 4 accumulators),
// block = 8 waves stacked in M = 256 M-rows x 32 N-cols per block.
// Ping/pong double buffer + manual 2x k-unroll: fragment loads are issued
// ~8 WMMAs ahead of consumption and buffers alternate with NO register
// copies (previous version burned 16 v_dual_mov_b32 per k-step on rotation).
// Requires K % 64 == 0 (holds: K = 1024 / 2048).
//
// Fragment lane mapping per CDNA5 ISA 7.12.2:
//   A  (16x32): lane r=lane&15 -> row; khalf=lane>>4: VGPR0-3 hold K=khalf*8+0..7,
//               VGPR4-7 hold K=16+khalf*8+0..7
//   Bt (as 32x16 B): lane -> column n=lane&15; lanes 0-15 K=0..7,16..23, 16-31 K=8..15,24..31
//   C/D: VGPR j -> row M = j + 8*khalf, col N = lane&15

__device__ __forceinline__ void load_fa(V16& f, const __bf16* __restrict__ row,
                                        int kk, int khalf) {
  f.u[0] = *(const uint4*)(row + kk + khalf * 8);
  f.u[1] = *(const uint4*)(row + kk + 16 + khalf * 8);
}
__device__ __forceinline__ void load_fb(V16& f, const __bf16* __restrict__ row,
                                        int kk, int khalf) {
  f.u[0] = *(const uint4*)(row + kk + khalf * 8);
  f.u[1] = *(const uint4*)(row + kk + 16 + khalf * 8);
}
__device__ __forceinline__ v8f wmma_bf16(const V16& a, const V16& b, v8f c) {
  return __builtin_amdgcn_wmma_f32_16x16x32_bf16(false, a.v, false, b.v,
                                                 (short)0, c, false, false);
}

__global__ __launch_bounds__(256) void wmma_gemm_kernel(const __bf16* __restrict__ A,
                                                        const __bf16* __restrict__ Bt,
                                                        float* __restrict__ C,
                                                        int M, int N, int K) {
  const int lane  = threadIdx.x & 31;
  const int wave  = threadIdx.x >> 5;
  const int r     = lane & 15;
  const int khalf = lane >> 4;

  const int mt = (blockIdx.y * 8 + wave) * 32;
  const int nt = blockIdx.x * 32;

  v8f acc00, acc01, acc10, acc11;
#pragma unroll
  for (int i = 0; i < 8; ++i) { acc00[i] = 0.f; acc01[i] = 0.f; acc10[i] = 0.f; acc11[i] = 0.f; }

  const __bf16* a0 = A  + (size_t)(mt + r) * K;
  const __bf16* a1 = A  + (size_t)(mt + 16 + r) * K;
  const __bf16* b0 = Bt + (size_t)(nt + r) * K;
  const __bf16* b1 = Bt + (size_t)(nt + 16 + r) * K;

  V16 pa0, pa1, pb0, pb1;    // ping: k = kk
  V16 qa0, qa1, qb0, qb1;    // pong: k = kk + 32
  load_fa(pa0, a0, 0, khalf);
  load_fa(pa1, a1, 0, khalf);
  load_fb(pb0, b0, 0, khalf);
  load_fb(pb1, b1, 0, khalf);
  load_fa(qa0, a0, 32, khalf);
  load_fa(qa1, a1, 32, khalf);
  load_fb(qb0, b0, 32, khalf);
  load_fb(qb1, b1, 32, khalf);

#pragma unroll 1
  for (int kk = 0; kk < K; kk += 64) {
    // consume ping (loads issued one 64-step earlier)
    acc00 = wmma_bf16(pa0, pb0, acc00);
    acc01 = wmma_bf16(pa0, pb1, acc01);
    acc10 = wmma_bf16(pa1, pb0, acc10);
    acc11 = wmma_bf16(pa1, pb1, acc11);
    if (kk + 64 < K) {                  // uniform branch: EXEC untouched
      load_fa(pa0, a0, kk + 64, khalf);
      load_fa(pa1, a1, kk + 64, khalf);
      load_fb(pb0, b0, kk + 64, khalf);
      load_fb(pb1, b1, kk + 64, khalf);
    }
    // consume pong
    acc00 = wmma_bf16(qa0, qb0, acc00);
    acc01 = wmma_bf16(qa0, qb1, acc01);
    acc10 = wmma_bf16(qa1, qb0, acc10);
    acc11 = wmma_bf16(qa1, qb1, acc11);
    if (kk + 96 < K) {
      load_fa(qa0, a0, kk + 96, khalf);
      load_fa(qa1, a1, kk + 96, khalf);
      load_fb(qb0, b0, kk + 96, khalf);
      load_fb(qb1, b1, kk + 96, khalf);
    }
  }

#pragma unroll
  for (int i = 0; i < 8; ++i) {
    const int m0 = mt + i + 8 * khalf;
    const int m1 = m0 + 16;
    C[(size_t)m0 * N + (nt + r)]      = acc00[i];
    C[(size_t)m0 * N + (nt + 16 + r)] = acc01[i];
    C[(size_t)m1 * N + (nt + r)]      = acc10[i];
    C[(size_t)m1 * N + (nt + 16 + r)] = acc11[i];
  }
}

// --------------------------- conv ------------------------------------------

__global__ __launch_bounds__(256) void conv_silu_kernel(const float* __restrict__ zx,
                                                        const float* __restrict__ w,
                                                        const float* __restrict__ bias,
                                                        float* __restrict__ out) {
  const int c  = blockIdx.x * 256 + threadIdx.x;      // channel in CONV_DIM
  const int ml = blockIdx.y;                          // b*SEQ + l
  if (c >= CONV_DIM) return;
  const int l = ml & (SEQ - 1);
  const int b = ml >> 11;
  float acc = bias[c];
#pragma unroll
  for (int k = 0; k < D_CONV; ++k) {
    const int ls = l + k - (D_CONV - 1);
    if (ls >= 0)
      acc += zx[((size_t)(b * SEQ + ls)) * DPROJ + D_INNER + c] * w[c * D_CONV + k];
  }
  out[(size_t)ml * CONV_DIM + c] = acc / (1.f + __expf(-acc));  // silu
}

// --------------------------- selective scan --------------------------------
// grid = (NHEADS, PSPLIT, BATCH) = 256 WGs. Each block owns 16 p-values x 128
// n-values of state; thread t: local p = t>>4, 8 n-values starting (t&15)*8.
// Per-step operands staged into LDS with b128 loads; softplus(dt) fused here
// (uniform per block -> scalar path). n-reduction via wave32 shuffles.
__global__ __launch_bounds__(256) void ssm_scan_kernel(const float* __restrict__ xbc,
                                                       const float* __restrict__ zx,
                                                       const float* __restrict__ dt_bias,
                                                       const float* __restrict__ A_log,
                                                       const float* __restrict__ Dv,
                                                       float* __restrict__ y) {
  const int h    = blockIdx.x;
  const int pblk = blockIdx.y;
  const int b    = blockIdx.z;
  const int t    = threadIdx.x;
  const int pl   = t >> 4;                 // 0..15
  const int p    = pblk * 16 + pl;
  const int n0   = (t & 15) * 8;

  const float A    = -__expf(A_log[h]);
  const float Dh   = Dv[h];
  const float bh   = dt_bias[h];

  __shared__ __align__(16) float sx[16];
  __shared__ __align__(16) float sB[D_STATE];
  __shared__ __align__(16) float sC[D_STATE];

  float hs[8];
#pragma unroll
  for (int j = 0; j < 8; ++j) hs[j] = 0.f;

  for (int l = 0; l < SEQ; ++l) {
    const size_t row = (size_t)(b * SEQ + l) * CONV_DIM;
    // stage x(16) + B(128) + C(128) floats with 68 b128 loads
    if (t < 4) {
      ((float4*)sx)[t] = *(const float4*)(xbc + row + h * HEADDIM + pblk * 16 + t * 4);
    } else if (t < 36) {
      ((float4*)sB)[t - 4] = *(const float4*)(xbc + row + D_INNER + (t - 4) * 4);
    } else if (t < 68) {
      ((float4*)sC)[t - 36] = *(const float4*)(xbc + row + D_INNER + D_STATE + (t - 36) * 4);
    }
    // softplus(dt_raw + bias): uniform per block
    const float draw  = zx[(size_t)(b * SEQ + l) * DPROJ + (DPROJ - NHEADS) + h] + bh;
    const float dtv   = (draw > 20.f) ? draw : log1pf(__expf(draw));
    const float decay = __expf(dtv * A);
    __syncthreads();

    const float xp   = sx[pl];
    const float coef = dtv * xp;
    float partial = 0.f;
#pragma unroll
    for (int j = 0; j < 8; ++j) {
      const float hn = hs[j] * decay + coef * sB[n0 + j];
      hs[j] = hn;
      partial += hn * sC[n0 + j];
    }
    partial += __shfl_xor(partial, 1, 32);
    partial += __shfl_xor(partial, 2, 32);
    partial += __shfl_xor(partial, 4, 32);
    partial += __shfl_xor(partial, 8, 32);
    if ((t & 15) == 0)
      y[(size_t)(b * SEQ + l) * D_INNER + h * HEADDIM + p] = partial + Dh * xp;
    __syncthreads();  // protect LDS before next step's staging
  }
}

// --------------------------- gate + RMS ------------------------------------

__global__ __launch_bounds__(256) void gate_rms_kernel(const float* __restrict__ y,
                                                       const float* __restrict__ zx,
                                                       const float* __restrict__ norm_w,
                                                       __bf16* __restrict__ ybf) {
  const int row = blockIdx.x;
  const int t   = threadIdx.x;
  float v[8];
  float ss = 0.f;
#pragma unroll
  for (int i = 0; i < 8; ++i) {
    const int e = t + i * 256;
    const float z = zx[(size_t)row * DPROJ + e];
    const float g = z / (1.f + __expf(-z));
    const float val = y[(size_t)row * D_INNER + e] * g;
    v[i] = val;
    ss += val * val;
  }
  __shared__ float red[256];
  red[t] = ss;
  __syncthreads();
  for (int s = 128; s > 0; s >>= 1) {
    if (t < s) red[t] += red[t + s];
    __syncthreads();
  }
  const float scale = rsqrtf(red[0] / (float)D_INNER + EPSV);
#pragma unroll
  for (int i = 0; i < 8; ++i) {
    const int e = t + i * 256;
    ybf[(size_t)row * D_INNER + e] = (__bf16)(v[i] * scale * norm_w[e]);
  }
}

__global__ __launch_bounds__(256) void final_rms_add_kernel(const float* __restrict__ og,
                                                            const float* __restrict__ x,
                                                            float* __restrict__ out) {
  const int row = blockIdx.x;
  const int t   = threadIdx.x;
  float v[4];
  float ss = 0.f;
#pragma unroll
  for (int i = 0; i < 4; ++i) {
    const int e = t + i * 256;
    const float a = og[(size_t)row * D_MODEL + e];
    v[i] = a;
    ss += a * a;
  }
  __shared__ float red[256];
  red[t] = ss;
  __syncthreads();
  for (int s = 128; s > 0; s >>= 1) {
    if (t < s) red[t] += red[t + s];
    __syncthreads();
  }
  const float scale = rsqrtf(red[0] / (float)D_MODEL + EPSV);
#pragma unroll
  for (int i = 0; i < 4; ++i) {
    const int e = t + i * 256;
    out[(size_t)row * D_MODEL + e] = v[i] * scale + x[(size_t)row * D_MODEL + e];
  }
}

// --------------------------- host launcher ---------------------------------

extern "C" void kernel_launch(void* const* d_in, const int* in_sizes, int n_in,
                              void* d_out, int out_size, void* d_ws, size_t ws_size,
                              hipStream_t stream) {
  (void)in_sizes; (void)n_in; (void)out_size; (void)ws_size;

  const float* x       = (const float*)d_in[0];
  const float* W_in    = (const float*)d_in[1];
  const float* conv_w  = (const float*)d_in[2];
  const float* conv_b  = (const float*)d_in[3];
  const float* dt_bias = (const float*)d_in[4];
  const float* A_log   = (const float*)d_in[5];
  const float* Dvec    = (const float*)d_in[6];
  const float* norm_w  = (const float*)d_in[7];
  const float* W_out   = (const float*)d_in[8];
  float* out = (float*)d_out;

  char* ws = (char*)d_ws;
  size_t off = 0;
  auto alloc = [&](size_t bytes) -> char* {
    char* p = ws + off;
    off += (bytes + 255) & ~(size_t)255;
    return p;
  };

  __bf16* xb    = (__bf16*)alloc((size_t)MROWS * D_MODEL * sizeof(__bf16));
  __bf16* WinT  = (__bf16*)alloc((size_t)DPROJ * D_MODEL * sizeof(__bf16));
  __bf16* WoutT = (__bf16*)alloc((size_t)D_MODEL * D_INNER * sizeof(__bf16));
  float*  zx    = (float*) alloc((size_t)MROWS * DPROJ * sizeof(float));
  float*  xbc   = (float*) alloc((size_t)MROWS * CONV_DIM * sizeof(float));
  float*  ybuf  = (float*) alloc((size_t)MROWS * D_INNER * sizeof(float));
  __bf16* ybf   = (__bf16*)alloc((size_t)MROWS * D_INNER * sizeof(__bf16));
  float*  og    = (float*) alloc((size_t)MROWS * D_MODEL * sizeof(float));

  // prep: bf16 convert + weight transposes
  cvt_bf16_kernel<<<(MROWS * D_MODEL + 255) / 256, 256, 0, stream>>>(x, xb, MROWS * D_MODEL);
  transpose_cvt_kernel<<<dim3(DPROJ / 32, D_MODEL / 32), 256, 0, stream>>>(W_in, WinT, D_MODEL, DPROJ);
  transpose_cvt_kernel<<<dim3(D_MODEL / 32, D_INNER / 32), 256, 0, stream>>>(W_out, WoutT, D_INNER, D_MODEL);

  // gemm1: zx[MROWS, DPROJ] = x @ W_in   (32-col strips: 4384/32 = 137 exact)
  wmma_gemm_kernel<<<dim3(DPROJ / 32, MROWS / 256), 256, 0, stream>>>(
      xb, WinT, zx, MROWS, DPROJ, D_MODEL);

  // conv + silu
  conv_silu_kernel<<<dim3(CONV_DIM / 256, MROWS), 256, 0, stream>>>(zx, conv_w, conv_b, xbc);

  // selective scan (softplus fused): 32 heads x 4 p-blocks x 2 batches = 256 WGs
  ssm_scan_kernel<<<dim3(NHEADS, PSPLIT, BATCH), 256, 0, stream>>>(
      xbc, zx, dt_bias, A_log, Dvec, ybuf);

  // gate + rms + convert to bf16 for gemm2
  gate_rms_kernel<<<MROWS, 256, 0, stream>>>(ybuf, zx, norm_w, ybf);

  // gemm2: og[MROWS, D_MODEL] = y @ W_out   (1024/32 = 32 strips)
  wmma_gemm_kernel<<<dim3(D_MODEL / 32, MROWS / 256), 256, 0, stream>>>(
      ybf, WoutT, og, MROWS, D_MODEL, D_INNER);

  // final rms + residual
  final_rms_add_kernel<<<MROWS, 256, 0, stream>>>(og, x, out);
}
